// Attention_9234179687676
// MI455X (gfx1250) — matrix operand order repaired and checked by reading, not compiled, not run on hardware
//
#include <hip/hip_runtime.h>

typedef __attribute__((ext_vector_type(16))) __bf16 v16bf;
typedef __attribute__((ext_vector_type(8)))  __bf16 v8bf;
typedef __attribute__((ext_vector_type(8)))  float  v8f;

#define DIMC    768
#define NHEADS  12
#define HDIM    64
#define BATCH   2
#define SEQ     4096
#define MTOT    (BATCH*SEQ)    /* 8192 */
#define QKVOUT  (3*DIMC)       /* 2304 */
#define SM_SCALE 0.125f        /* 64^-0.5 */

#define ZERO8F (v8f){0.f,0.f,0.f,0.f,0.f,0.f,0.f,0.f}

// ---- fragment loaders matched to CDNA5 WMMA VGPR layouts ----
// A (16x32 bf16): lane holds 8 contiguous K at base, 8 more at base+16
static __device__ inline v16bf load_frag_a(const __bf16* p) {
  v16bf a;
  *reinterpret_cast<v8bf*>(&a)       = *reinterpret_cast<const v8bf*>(p);
  *(reinterpret_cast<v8bf*>(&a) + 1) = *reinterpret_cast<const v8bf*>(p + 16);
  return a;
}
// B (32x16 bf16, N-major source): lane holds 8 contiguous K at base, 8 more at base+16
static __device__ inline v16bf load_frag_b(const __bf16* p) {
  v16bf b;
  *reinterpret_cast<v8bf*>(&b)       = *reinterpret_cast<const v8bf*>(p);
  *(reinterpret_cast<v8bf*>(&b) + 1) = *reinterpret_cast<const v8bf*>(p + 16);
  return b;
}

static __device__ inline float red_max16(float v) {
  v = fmaxf(v, __shfl_xor(v, 1, 32));
  v = fmaxf(v, __shfl_xor(v, 2, 32));
  v = fmaxf(v, __shfl_xor(v, 4, 32));
  v = fmaxf(v, __shfl_xor(v, 8, 32));
  return v;
}
static __device__ inline float red_sum16(float v) {
  v += __shfl_xor(v, 1, 32);
  v += __shfl_xor(v, 2, 32);
  v += __shfl_xor(v, 4, 32);
  v += __shfl_xor(v, 8, 32);
  return v;
}

// ---------------- f32 -> bf16 conversion ----------------
__global__ void cvt_bf16_kernel(const float* __restrict__ in,
                                __bf16* __restrict__ out, int n) {
  int i = blockIdx.x * blockDim.x + threadIdx.x;
  if (i < n) out[i] = (__bf16)in[i];
}

// ---------------- QKV GEMM: [8192x768] x W^T[768x2304] ----------------
// Wave computes 32x64 tile (2 A-frags x 4 B-tiles -> 8 WMMA / k-step).
// Epilogue scatters Q,K -> [b,h,n,d], V -> [b,h,d,n].
__global__ __launch_bounds__(256) void gemm_qkv_kernel(
    const __bf16* __restrict__ A, const __bf16* __restrict__ W,
    __bf16* __restrict__ qb, __bf16* __restrict__ kb, __bf16* __restrict__ vtb) {
  const int warp = threadIdx.x >> 5;
  const int lane = threadIdx.x & 31;
  const int hlf = lane >> 4, lm = lane & 15;
  const int NCG = QKVOUT / 64;                 // 36 column groups
  int wid = blockIdx.x * 8 + warp;
  int m0 = (wid / NCG) * 32;
  int n0 = (wid % NCG) * 64;

  v8f c[2][4];
#pragma unroll
  for (int mh = 0; mh < 2; ++mh)
#pragma unroll
    for (int j = 0; j < 4; ++j) c[mh][j] = ZERO8F;

  for (int k0 = 0; k0 < DIMC; k0 += 32) {
    v16bf a0 = load_frag_a(A + (size_t)(m0 + lm) * DIMC + k0 + hlf * 8);
    v16bf a1 = load_frag_a(A + (size_t)(m0 + 16 + lm) * DIMC + k0 + hlf * 8);
#pragma unroll
    for (int j = 0; j < 4; ++j) {
      v16bf b = load_frag_b(W + (size_t)(n0 + j * 16 + lm) * DIMC + k0 + hlf * 8);
      c[0][j] = __builtin_amdgcn_wmma_f32_16x16x32_bf16(false, a0, false, b,
                                                        (short)0, c[0][j], false, false);
      c[1][j] = __builtin_amdgcn_wmma_f32_16x16x32_bf16(false, a1, false, b,
                                                        (short)0, c[1][j], false, false);
    }
  }
#pragma unroll
  for (int j = 0; j < 4; ++j) {
    int o = n0 + j * 16 + lm;
    int which = o / DIMC;
    int cc = o % DIMC;
    int h = cc >> 6, d = cc & 63;
#pragma unroll
    for (int mh = 0; mh < 2; ++mh) {
#pragma unroll
      for (int r = 0; r < 8; ++r) {
        int gm = m0 + mh * 16 + r + hlf * 8;
        int b_ = gm >> 12, n = gm & 4095;
        __bf16 bv = (__bf16)c[mh][j][r];
        size_t bh = (size_t)(b_ * NHEADS + h);
        if (which == 0)      qb [(bh * SEQ + n) * HDIM + d] = bv;
        else if (which == 1) kb [(bh * SEQ + n) * HDIM + d] = bv;
        else                 vtb[(bh * HDIM + d) * SEQ + n] = bv;   // V transposed
      }
    }
  }
}

// ---------------- Flash attention: 32 query rows per wave ----------------
__global__ __launch_bounds__(256) void flash_attn_kernel(
    const __bf16* __restrict__ q, const __bf16* __restrict__ kk,
    const __bf16* __restrict__ vt, __bf16* __restrict__ aatt) {
  __shared__ __bf16 smem[8 * 32 * 64];          // 4 KB per wave, 32 KB/block
  const int warp = threadIdx.x >> 5;
  const int lane = threadIdx.x & 31;
  const int hlf = lane >> 4, lm = lane & 15;
  int wj = blockIdx.x * 8 + warp;
  int bh = wj / 128;                             // 128 q-tiles (of 32) per (b,h)
  int q0 = (wj % 128) * 32;
  __bf16* pl = smem + warp * (32 * 64);

  v16bf qa[2][2];
#pragma unroll
  for (int mh = 0; mh < 2; ++mh) {
    const __bf16* qbase = q + ((size_t)bh * SEQ + q0 + mh * 16 + lm) * HDIM;
    qa[mh][0] = load_frag_a(qbase + 0  + hlf * 8);
    qa[mh][1] = load_frag_a(qbase + 32 + hlf * 8);
  }

  v8f oacc[2][4];
  float mr[2][8], lr[2][8];
#pragma unroll
  for (int mh = 0; mh < 2; ++mh) {
#pragma unroll
    for (int d = 0; d < 4; ++d) oacc[mh][d] = ZERO8F;
#pragma unroll
    for (int r = 0; r < 8; ++r) { mr[mh][r] = -__builtin_inff(); lr[mh][r] = 0.f; }
  }

  for (int kv0 = 0; kv0 < SEQ; kv0 += 64) {
    // ---- S = Q K^T (32 x 64 tile); K-fragments reused by both row halves ----
    v8f s[2][4];
#pragma unroll
    for (int j = 0; j < 4; ++j) {
      const __bf16* kp = kk + ((size_t)bh * SEQ + kv0 + j * 16 + lm) * HDIM;
      v16bf b0 = load_frag_b(kp + 0  + hlf * 8);
      v16bf b1 = load_frag_b(kp + 32 + hlf * 8);
#pragma unroll
      for (int mh = 0; mh < 2; ++mh) {
        v8f acc = ZERO8F;
        acc = __builtin_amdgcn_wmma_f32_16x16x32_bf16(false, qa[mh][0], false, b0,
                                                      (short)0, acc, false, false);
        acc = __builtin_amdgcn_wmma_f32_16x16x32_bf16(false, qa[mh][1], false, b1,
                                                      (short)0, acc, false, false);
        s[mh][j] = acc;
      }
    }
    // ---- online softmax; write P (bf16) to LDS in row-major ----
#pragma unroll
    for (int mh = 0; mh < 2; ++mh) {
#pragma unroll
      for (int r = 0; r < 8; ++r) {
        float s0 = s[mh][0][r] * SM_SCALE, s1 = s[mh][1][r] * SM_SCALE;
        float s2 = s[mh][2][r] * SM_SCALE, s3 = s[mh][3][r] * SM_SCALE;
        float tmax = red_max16(fmaxf(fmaxf(s0, s1), fmaxf(s2, s3)));
        float mnew = fmaxf(mr[mh][r], tmax);
        float alpha = __expf(mr[mh][r] - mnew);
        float p0 = __expf(s0 - mnew), p1 = __expf(s1 - mnew);
        float p2 = __expf(s2 - mnew), p3 = __expf(s3 - mnew);
        int row = mh * 16 + r + hlf * 8;
        pl[row * 64 +  0 + lm] = (__bf16)p0;
        pl[row * 64 + 16 + lm] = (__bf16)p1;
        pl[row * 64 + 32 + lm] = (__bf16)p2;
        pl[row * 64 + 48 + lm] = (__bf16)p3;
        lr[mh][r] = lr[mh][r] * alpha + red_sum16(p0 + p1 + p2 + p3);
        mr[mh][r] = mnew;
#pragma unroll
        for (int d = 0; d < 4; ++d) oacc[mh][d][r] *= alpha;
      }
    }
    asm volatile("s_wait_dscnt 0x0" ::: "memory");   // LDS bounce: stores -> loads
    // ---- O += P V; V-fragments reused by both row halves ----
    v16bf pa[2][2];
#pragma unroll
    for (int mh = 0; mh < 2; ++mh) {
      const __bf16* pbase = pl + (mh * 16 + lm) * 64;
      pa[mh][0] = load_frag_a(pbase + 0  + hlf * 8);
      pa[mh][1] = load_frag_a(pbase + 32 + hlf * 8);
    }
#pragma unroll
    for (int dt = 0; dt < 4; ++dt) {
      const __bf16* vp = vt + ((size_t)bh * HDIM + dt * 16 + lm) * SEQ + kv0;
      v16bf b0 = load_frag_b(vp + 0  + hlf * 8);
      v16bf b1 = load_frag_b(vp + 32 + hlf * 8);
#pragma unroll
      for (int mh = 0; mh < 2; ++mh) {
        oacc[mh][dt] = __builtin_amdgcn_wmma_f32_16x16x32_bf16(false, pa[mh][0], false, b0,
                                                               (short)0, oacc[mh][dt], false, false);
        oacc[mh][dt] = __builtin_amdgcn_wmma_f32_16x16x32_bf16(false, pa[mh][1], false, b1,
                                                               (short)0, oacc[mh][dt], false, false);
      }
    }
  }
  // ---- normalize + write [b,n,c] bf16 for the proj GEMM ----
  int b = bh / NHEADS, h = bh % NHEADS;
#pragma unroll
  for (int mh = 0; mh < 2; ++mh) {
#pragma unroll
    for (int r = 0; r < 8; ++r) {
      float inv = 1.f / lr[mh][r];
      int n = q0 + mh * 16 + r + hlf * 8;
#pragma unroll
      for (int dt = 0; dt < 4; ++dt)
        aatt[((size_t)b * SEQ + n) * DIMC + h * HDIM + dt * 16 + lm] =
            (__bf16)(oacc[mh][dt][r] * inv);
    }
  }
}

// ---------------- Proj GEMM: [8192x768] x Wp^T[768x768] + bias ----------------
__global__ __launch_bounds__(256) void gemm_proj_kernel(
    const __bf16* __restrict__ A, const __bf16* __restrict__ W,
    const float* __restrict__ bias, float* __restrict__ out) {
  const int warp = threadIdx.x >> 5;
  const int lane = threadIdx.x & 31;
  const int hlf = lane >> 4, lm = lane & 15;
  const int NCG = DIMC / 64;                     // 12
  int wid = blockIdx.x * 8 + warp;
  int m0 = (wid / NCG) * 32;
  int n0 = (wid % NCG) * 64;

  v8f c[2][4];
#pragma unroll
  for (int mh = 0; mh < 2; ++mh)
#pragma unroll
    for (int j = 0; j < 4; ++j) c[mh][j] = ZERO8F;

  for (int k0 = 0; k0 < DIMC; k0 += 32) {
    v16bf a0 = load_frag_a(A + (size_t)(m0 + lm) * DIMC + k0 + hlf * 8);
    v16bf a1 = load_frag_a(A + (size_t)(m0 + 16 + lm) * DIMC + k0 + hlf * 8);
#pragma unroll
    for (int j = 0; j < 4; ++j) {
      v16bf b = load_frag_b(W + (size_t)(n0 + j * 16 + lm) * DIMC + k0 + hlf * 8);
      c[0][j] = __builtin_amdgcn_wmma_f32_16x16x32_bf16(false, a0, false, b,
                                                        (short)0, c[0][j], false, false);
      c[1][j] = __builtin_amdgcn_wmma_f32_16x16x32_bf16(false, a1, false, b,
                                                        (short)0, c[1][j], false, false);
    }
  }
#pragma unroll
  for (int j = 0; j < 4; ++j) {
    int o = n0 + j * 16 + lm;
    float bv = bias[o];
#pragma unroll
    for (int mh = 0; mh < 2; ++mh) {
#pragma unroll
      for (int r = 0; r < 8; ++r) {
        int gm = m0 + mh * 16 + r + hlf * 8;
        out[(size_t)gm * DIMC + o] = c[mh][j][r] + bv;
      }
    }
  }
}

// ---------------- host launcher ----------------
extern "C" void kernel_launch(void* const* d_in, const int* in_sizes, int n_in,
                              void* d_out, int out_size, void* d_ws, size_t ws_size,
                              hipStream_t stream) {
  const float* x      = (const float*)d_in[0];   // [2,4096,768]
  const float* w_qkv  = (const float*)d_in[1];   // [2304,768]
  const float* w_proj = (const float*)d_in[2];   // [768,768]
  const float* b_proj = (const float*)d_in[3];   // [768]
  float* out = (float*)d_out;

  const size_t XE  = (size_t)MTOT * DIMC;        // 6291456
  const size_t WQE = (size_t)QKVOUT * DIMC;      // 1769472
  const size_t WPE = (size_t)DIMC * DIMC;        // 589824
  char* ws = (char*)d_ws;
  __bf16* xb    = (__bf16*)ws;                  ws += XE  * 2;
  __bf16* wqkvb = (__bf16*)ws;                  ws += WQE * 2;
  __bf16* wprojb= (__bf16*)ws;                  ws += WPE * 2;
  __bf16* qb    = (__bf16*)ws;                  ws += XE  * 2;   // [b,h,n,d]
  __bf16* kb    = (__bf16*)ws;                  ws += XE  * 2;   // [b,h,n,d]
  __bf16* vtb   = (__bf16*)ws;                  ws += XE  * 2;   // [b,h,d,n]
  __bf16* aatt  = (__bf16*)ws;                  ws += XE  * 2;   // [b,n,c]

  cvt_bf16_kernel<<<(XE  + 255) / 256, 256, 0, stream>>>(x,      xb,     (int)XE);
  cvt_bf16_kernel<<<(WQE + 255) / 256, 256, 0, stream>>>(w_qkv,  wqkvb,  (int)WQE);
  cvt_bf16_kernel<<<(WPE + 255) / 256, 256, 0, stream>>>(w_proj, wprojb, (int)WPE);

  // (8192/32) row tiles * (2304/64) col groups = 9216 waves / 8 per block
  gemm_qkv_kernel<<<1152, 256, 0, stream>>>(xb, wqkvb, qb, kb, vtb);

  // 24 (b,h) * 128 q-tiles = 3072 waves / 8 per block
  flash_attn_kernel<<<384, 256, 0, stream>>>(qb, kb, vtb, aatt);

  // (8192/32) * (768/64) = 3072 waves / 8 per block
  gemm_proj_kernel<<<384, 256, 0, stream>>>(aatt, wprojb, b_proj, out);
}
